// DiagonalSSM_62302795596590
// MI455X (gfx1250) — matrix operand order repaired and checked by reading, not compiled, not run on hardware
//
#include <hip/hip_runtime.h>
#include <hip/hip_bf16.h>
#include <math.h>

// ---------------- problem constants (from reference) ----------------
#define NBATCH 4
#define SEQ    2048
#define TTOK   (NBATCH * SEQ)   // 8192 rows
#define DIM    1024
#define NSTATE 64
#define HDIM   4096
#define NLAYER 2
#define NCLS   1000
#define D2     2048

// ---------------- WMMA types ----------------
typedef __attribute__((ext_vector_type(16))) _Float16 v16h;
typedef __attribute__((ext_vector_type(8)))  float    v8f;
typedef __attribute__((ext_vector_type(4)))  int      v4i;

union Frag { _Float16 e[16]; v16h v; };

#define AS1 __attribute__((address_space(1)))
#define AS3 __attribute__((address_space(3)))

// ---- CDNA5 async global->LDS copy (ASYNCcnt-tracked DMA), with fallback ----
#if defined(__AMDGCN__) && __has_builtin(__builtin_amdgcn_global_load_async_to_lds_b128) && __has_builtin(__builtin_amdgcn_s_wait_asynccnt)
#define HAVE_ASYNC 1
#else
#define HAVE_ASYNC 0
#endif

__device__ __forceinline__ void async_cp16(const float* g, float* l) {
#if HAVE_ASYNC
    __builtin_amdgcn_global_load_async_to_lds_b128(
        (AS1 v4i*)(AS1 const v4i*)g, (AS3 v4i*)l, 0, 0);
#else
    *(float4*)l = *(const float4*)g;
#endif
}
__device__ __forceinline__ void async_wait() {
#if HAVE_ASYNC
    __builtin_amdgcn_s_wait_asynccnt(0);
#endif
}

__device__ __forceinline__ float sigm(float x) { return 1.0f / (1.0f + __expf(-x)); }

// epilogue ops
#define OP_NONE         0
#define OP_BIAS_SIGMOID 1
#define OP_SIGMOID      2
#define OP_BIAS_GELU    3
#define OP_BIAS_ADD     4
#define OP_RES_ADD      5   // C += acc
#define OP_MUL_SILU     6   // C  = C * silu(acc)

// ---------------- WMMA GEMM: C(MxN) = epilogue(A(MxK,row) @ W(KxN,row)) ----------------
#define BM 128
#define BN 64
#define BK 32
#define LDA_PF 36   // A LDS row stride (floats); 36*4B keeps b128 frag reads at bank floor

// XOR chunk swizzle for the B tile [k][n] f32: keeps DMA dst 16B-aligned while
// making per-column fragment gathers single-pass across all 32 lanes.
__device__ __forceinline__ int bswz(int kk, int n) {
    int s = (kk & 15) ^ ((kk >> 4) << 2);
    return kk * 64 + ((((n >> 2) ^ s) & 15) << 2) + (n & 3);
}

template <int OP, bool FULLN>
__global__ __launch_bounds__(256)
void wmma_gemm(const float* __restrict__ A, const float* __restrict__ W,
               float* __restrict__ C, const float* __restrict__ bias,
               int M, int N, int K, int col_base)
{
    __shared__ __align__(16) float sA[2][BM * LDA_PF];
    __shared__ __align__(16) float sB[2][BK * BN];

    const int tid  = threadIdx.x;
    const int lane = tid & 31;
    const int wave = tid >> 5;      // 8 waves
    const int wm   = wave >> 1;     // 0..3 (M)
    const int wn   = wave & 1;      // 0..1 (N)
    const int row0 = blockIdx.x * BM;
    const int col0 = col_base + blockIdx.y * BN;
    const int mrow = lane & 15;
    const int hi   = lane >> 4;
    const int kba  = 8 * hi;        // A frag: halves = K{kba..kba+7, kba+16..kba+23}
    const int kbb  = 8 * hi;        // B frag: halves = K{kbb..kbb+7, kbb+16..kbb+23}
    (void)M;

    v8f zero = {0.f,0.f,0.f,0.f,0.f,0.f,0.f,0.f};
    v8f acc[2][2];
    acc[0][0] = zero; acc[0][1] = zero; acc[1][0] = zero; acc[1][1] = zero;

    // ---- stagers: raw f32 tiles, 16B per thread-chunk ----
    auto stageA = [&](const float* Ak, float* buf) {
        #pragma unroll
        for (int i = 0; i < 4; ++i) {           // 128x32 f32 = 16 KB
            int e = i * 1024 + tid * 4;
            int r = e >> 5, c = e & 31;
            async_cp16(Ak + (size_t)r * K + c, buf + r * LDA_PF + c);
        }
    };
    auto stageB = [&](const float* Wk, float* buf) {
        #pragma unroll
        for (int i = 0; i < 2; ++i) {           // 32x64 f32 = 8 KB
            int e = i * 1024 + tid * 4;
            int kk = e >> 6, n = e & 63;
            async_cp16(Wk + (size_t)kk * N + n, buf + bswz(kk, n));
        }
    };
    auto stageB_guard = [&](const float* Wk, float* buf) {   // tail N-block, scalar+zero-fill
        #pragma unroll
        for (int i = 0; i < 2; ++i) {
            int e = i * 1024 + tid * 4;
            int kk = e >> 6, n = e & 63;
            #pragma unroll
            for (int j = 0; j < 4; ++j) {
                int gc = col0 + n + j;
                buf[bswz(kk, n + j)] = (gc < N) ? Wk[(size_t)kk * N + n + j] : 0.0f;
            }
        }
    };

    // ---- compute one 32-deep K step from staged f32 tiles (cvt in-regs) ----
    auto computeTile = [&](const float* bufA, const float* bufB) {
        Frag a[2], b[2];
        #pragma unroll
        for (int mt = 0; mt < 2; ++mt) {
            const float* p = bufA + (wm * 32 + mt * 16 + mrow) * LDA_PF + kba;
            float4 x0 = *(const float4*)(p);
            float4 x1 = *(const float4*)(p + 4);
            float4 x2 = *(const float4*)(p + 16);
            float4 x3 = *(const float4*)(p + 20);
            a[mt].e[0]=(_Float16)x0.x; a[mt].e[1]=(_Float16)x0.y; a[mt].e[2]=(_Float16)x0.z; a[mt].e[3]=(_Float16)x0.w;
            a[mt].e[4]=(_Float16)x1.x; a[mt].e[5]=(_Float16)x1.y; a[mt].e[6]=(_Float16)x1.z; a[mt].e[7]=(_Float16)x1.w;
            a[mt].e[8]=(_Float16)x2.x; a[mt].e[9]=(_Float16)x2.y; a[mt].e[10]=(_Float16)x2.z; a[mt].e[11]=(_Float16)x2.w;
            a[mt].e[12]=(_Float16)x3.x; a[mt].e[13]=(_Float16)x3.y; a[mt].e[14]=(_Float16)x3.z; a[mt].e[15]=(_Float16)x3.w;
        }
        #pragma unroll
        for (int nt = 0; nt < 2; ++nt) {
            int ncg = wn * 32 + nt * 16 + mrow;
            #pragma unroll
            for (int k = 0; k < 16; ++k)
                b[nt].e[k] = (_Float16)bufB[bswz(kbb + k + ((k >> 3) << 3), ncg)];
        }
        #pragma unroll
        for (int mt = 0; mt < 2; ++mt)
            #pragma unroll
            for (int nt = 0; nt < 2; ++nt)
                acc[mt][nt] = __builtin_amdgcn_wmma_f32_16x16x32_f16(
                    false, a[mt].v, false, b[nt].v, (short)0, acc[mt][nt], false, false);
    };

    const int NT = K / BK;
    if constexpr (FULLN) {
        // ---- async ping-pong: copy tile t+1 while computing tile t ----
        stageA(A + (size_t)row0 * K, &sA[0][0]);
        stageB(W + col0, &sB[0][0]);
        for (int t = 0; t < NT; ++t) {
            async_wait();
            __syncthreads();
            if (t + 1 < NT) {
                int kn = (t + 1) * BK;
                stageA(A + (size_t)row0 * K + kn, &sA[(t + 1) & 1][0]);
                stageB(W + (size_t)kn * N + col0, &sB[(t + 1) & 1][0]);
            }
            computeTile(&sA[t & 1][0], &sB[t & 1][0]);
        }
    } else {
        for (int t = 0; t < NT; ++t) {
            int kn = t * BK;
            __syncthreads();
            stageA(A + (size_t)row0 * K + kn, &sA[0][0]);
            stageB_guard(W + (size_t)kn * N + col0, &sB[0][0]);
            async_wait();
            __syncthreads();
            computeTile(&sA[0][0], &sB[0][0]);
        }
    }

    // ---- epilogue: C/D layout M = r + 8*(lane>=16), N = lane&15 ----
    const int mrow8 = 8 * hi;
    #pragma unroll
    for (int mt = 0; mt < 2; ++mt)
        #pragma unroll
        for (int nt = 0; nt < 2; ++nt)
            #pragma unroll
            for (int r = 0; r < 8; ++r) {
                int row = row0 + wm * 32 + mt * 16 + mrow8 + r;
                int col = col0 + wn * 32 + nt * 16 + mrow;
                if (!FULLN && col >= N) continue;
                size_t idx = (size_t)row * N + col;
                float v = acc[mt][nt][r];
                if constexpr (OP == OP_BIAS_SIGMOID) v = sigm(v + bias[col]);
                else if constexpr (OP == OP_SIGMOID) v = sigm(v);
                else if constexpr (OP == OP_BIAS_GELU) {
                    float t = v + bias[col];
                    v = 0.5f * t * (1.0f + erff(t * 0.70710678118654752f));
                }
                else if constexpr (OP == OP_BIAS_ADD) v = v + bias[col];
                else if constexpr (OP == OP_RES_ADD)  v = C[idx] + v;
                else if constexpr (OP == OP_MUL_SILU) v = C[idx] * (v * sigm(v));
                C[idx] = v;
            }
}

// ---------------- embed: x = tok_emb[token] + pos_emb[s] ----------------
__global__ __launch_bounds__(256)
void embed_k(const int* __restrict__ tokens, const float* __restrict__ tok_emb,
             const float* __restrict__ pos_emb, float* __restrict__ x)
{
    int row = blockIdx.x;
    int s   = row % SEQ;
    int tok = tokens[row];
    const float* te = tok_emb + (size_t)tok * DIM;
    const float* pe = pos_emb + (size_t)s * DIM;
    float* xr = x + (size_t)row * DIM;
    for (int d = threadIdx.x; d < DIM; d += 256) xr[d] = te[d] + pe[d];
}

// ---------------- rmsnorm: out = x * rsqrt(mean(x^2)+eps) * w ----------------
__global__ __launch_bounds__(256)
void rmsnorm_k(const float* __restrict__ x, const float* __restrict__ w,
               float* __restrict__ out)
{
    __shared__ float red[256];
    int row = blockIdx.x, tid = threadIdx.x;
    const float* xr = x + (size_t)row * DIM;
    float s = 0.0f;
    #pragma unroll
    for (int i = 0; i < DIM / 256; ++i) { float v = xr[tid + i * 256]; s += v * v; }
    red[tid] = s; __syncthreads();
    for (int off = 128; off > 0; off >>= 1) {
        if (tid < off) red[tid] += red[tid + off];
        __syncthreads();
    }
    float scale = rsqrtf(red[0] * (1.0f / DIM) + 1e-6f);
    float* o = out + (size_t)row * DIM;
    #pragma unroll
    for (int i = 0; i < DIM / 256; ++i) { int d = tid + i * 256; o[d] = xr[d] * scale * w[d]; }
}

// ---------------- diagonal SSM scan: h_t = m_t*(lam_t*h_{t-1} + u_t) ----------------
__global__ void scan_k(const float* __restrict__ lam, const float* __restrict__ u,
                       const float* __restrict__ mask, float* __restrict__ h)
{
    int b = blockIdx.x, n = threadIdx.x;   // 64 channels / batch
    float hv = 0.0f;
    for (int s = 0; s < SEQ; ++s) {
        size_t i = ((size_t)(b * SEQ + s)) * NSTATE + n;
        float m = mask[b * SEQ + s];
        hv = m * (lam[i] * hv + u[i]);
        h[i] = hv;
    }
}

// ---------------- gated out-norm + residual: x += rmsnorm(gate*y, w) ----------------
__global__ __launch_bounds__(256)
void gateout_k(float* __restrict__ x, const float* __restrict__ gate,
               const float* __restrict__ y, const float* __restrict__ w)
{
    __shared__ float red[256];
    int row = blockIdx.x, tid = threadIdx.x;
    size_t base = (size_t)row * DIM;
    float t[DIM / 256];
    float s = 0.0f;
    #pragma unroll
    for (int i = 0; i < DIM / 256; ++i) {
        int d = tid + i * 256;
        t[i] = gate[base + d] * y[base + d];
        s += t[i] * t[i];
    }
    red[tid] = s; __syncthreads();
    for (int off = 128; off > 0; off >>= 1) {
        if (tid < off) red[tid] += red[tid + off];
        __syncthreads();
    }
    float scale = rsqrtf(red[0] * (1.0f / DIM) + 1e-6f);
    #pragma unroll
    for (int i = 0; i < DIM / 256; ++i) {
        int d = tid + i * 256;
        x[base + d] += t[i] * scale * w[d];
    }
}

// ---------------- launch ----------------
extern "C" void kernel_launch(void* const* d_in, const int* in_sizes, int n_in,
                              void* d_out, int out_size, void* d_ws, size_t ws_size,
                              hipStream_t stream)
{
    const int*   tokens   = (const int*)  d_in[0];
    const float* mask     = (const float*)d_in[1];
    const float* tok_emb  = (const float*)d_in[2];
    const float* pos_emb  = (const float*)d_in[3];
    const float* ssm_nw   = (const float*)d_in[4];
    const float* lam_w    = (const float*)d_in[5];
    const float* lam_b    = (const float*)d_in[6];
    const float* B_w      = (const float*)d_in[7];
    const float* C_w      = (const float*)d_in[8];
    const float* gate_w   = (const float*)d_in[9];
    const float* out_nw   = (const float*)d_in[10];
    const float* ffn_nw   = (const float*)d_in[11];
    const float* w1       = (const float*)d_in[12];
    const float* w3       = (const float*)d_in[13];
    const float* w2       = (const float*)d_in[14];
    const float* final_nw = (const float*)d_in[15];
    const float* fc1_w    = (const float*)d_in[16];
    const float* fc1_b    = (const float*)d_in[17];
    const float* fc2_w    = (const float*)d_in[18];
    const float* fc2_b    = (const float*)d_in[19];
    float* out = (float*)d_out;

    // workspace carve-up (floats)
    float* ws = (float*)d_ws;
    size_t off = 0;
    float* X    = ws + off; off += (size_t)TTOK * DIM;
    float* XN   = ws + off; off += (size_t)TTOK * DIM;
    float* Y    = ws + off; off += (size_t)TTOK * DIM;
    float* GATE = ws + off; off += (size_t)TTOK * DIM;
    float* LAM  = ws + off; off += (size_t)TTOK * NSTATE;
    float* U    = ws + off; off += (size_t)TTOK * NSTATE;
    float* HS   = ws + off; off += (size_t)TTOK * NSTATE;
    float* G    = ws + off; off += (size_t)TTOK * HDIM;   // FFN hidden; reused as fc1 hidden
    (void)off; (void)ws_size; (void)in_sizes; (void)n_in; (void)out_size;

    const dim3 blk(256);
    const int MB = TTOK / BM;   // 64 M-tiles

    embed_k<<<TTOK, blk, 0, stream>>>(tokens, tok_emb, pos_emb, X);

    for (int l = 0; l < NLAYER; ++l) {
        rmsnorm_k<<<TTOK, blk, 0, stream>>>(X, ssm_nw + (size_t)l * DIM, XN);

        wmma_gemm<OP_BIAS_SIGMOID, true><<<dim3(MB, NSTATE / BN), blk, 0, stream>>>(
            XN, lam_w + (size_t)l * DIM * NSTATE, LAM, lam_b + (size_t)l * NSTATE,
            TTOK, NSTATE, DIM, 0);
        wmma_gemm<OP_NONE, true><<<dim3(MB, NSTATE / BN), blk, 0, stream>>>(
            XN, B_w + (size_t)l * DIM * NSTATE, U, nullptr, TTOK, NSTATE, DIM, 0);

        scan_k<<<NBATCH, NSTATE, 0, stream>>>(LAM, U, mask, HS);

        wmma_gemm<OP_NONE, true><<<dim3(MB, DIM / BN), blk, 0, stream>>>(
            HS, C_w + (size_t)l * NSTATE * DIM, Y, nullptr, TTOK, DIM, NSTATE, 0);
        wmma_gemm<OP_SIGMOID, true><<<dim3(MB, DIM / BN), blk, 0, stream>>>(
            XN, gate_w + (size_t)l * DIM * DIM, GATE, nullptr, TTOK, DIM, DIM, 0);

        gateout_k<<<TTOK, blk, 0, stream>>>(X, GATE, Y, out_nw + (size_t)l * DIM);

        rmsnorm_k<<<TTOK, blk, 0, stream>>>(X, ffn_nw + (size_t)l * DIM, XN);

        wmma_gemm<OP_NONE, true><<<dim3(MB, HDIM / BN), blk, 0, stream>>>(
            XN, w3 + (size_t)l * DIM * HDIM, G, nullptr, TTOK, HDIM, DIM, 0);
        wmma_gemm<OP_MUL_SILU, true><<<dim3(MB, HDIM / BN), blk, 0, stream>>>(
            XN, w1 + (size_t)l * DIM * HDIM, G, nullptr, TTOK, HDIM, DIM, 0);
        wmma_gemm<OP_RES_ADD, true><<<dim3(MB, DIM / BN), blk, 0, stream>>>(
            G, w2 + (size_t)l * HDIM * DIM, X, nullptr, TTOK, DIM, HDIM, 0);
    }

    rmsnorm_k<<<TTOK, blk, 0, stream>>>(X, final_nw, XN);
    wmma_gemm<OP_BIAS_GELU, true><<<dim3(MB, D2 / BN), blk, 0, stream>>>(
        XN, fc1_w, G, fc1_b, TTOK, D2, DIM, 0);
    // fc2: 15 full 64-col blocks + guarded tail block (cols 960..999)
    wmma_gemm<OP_BIAS_ADD, true><<<dim3(MB, NCLS / BN), blk, 0, stream>>>(
        G, fc2_w, out, fc2_b, TTOK, NCLS, D2, 0);
    wmma_gemm<OP_BIAS_ADD, false><<<dim3(MB, 1), blk, 0, stream>>>(
        G, fc2_w, out, fc2_b, TTOK, NCLS, D2, (NCLS / BN) * BN);
}
